// Attention_12945031430219
// MI455X (gfx1250) — compile-verified, matrix operand order checked by reading, not run on hardware
//
#include <hip/hip_runtime.h>
#include <hip/hip_bf16.h>
#include <stdint.h>

typedef __attribute__((ext_vector_type(16))) _Float16 v16h;
typedef __attribute__((ext_vector_type(8)))  _Float16 v8h;
typedef __attribute__((ext_vector_type(8)))  float    v8f;

union Frag16 { v16h v; v8h h[2]; };
union Acc8   { v8f  v; float f[8]; };

#define BATCH 64
#define SEQ   197
#define CH    768
#define HEADS 12
#define HD    64
#define NP    224            // SEQ padded to multiple of 32 (7*32, 14*16)
#define BN    (BATCH*SEQ)    // 12608 = 788*16 exactly
#define MT    (BN/16)        // 788 row tiles (788 = 4*197)

// ---- WMMA helper: D = A(16x32 f16) * B(32x16 f16) + C(16x16 f32) ----
__device__ __forceinline__ v8f wmma16(v16h a, v16h b, v8f c) {
  return __builtin_amdgcn_wmma_f32_16x16x32_f16(false, a, false, b, (short)0, c,
                                                false, false);
}

// A-fragment (16x32, row-major source, caller passes pointer at its own row):
// lane r (0..15), hi = lane>>4. elements 0..7 -> k = kb + hi*8 + e
//                               elements 8..15 -> k = kb + 16 + hi*8 + e
// Same mapping serves the B-fragment when B is available as B^T row-major.
__device__ __forceinline__ v16h load_frag(const _Float16* rowptr, int kb, int hi) {
  const _Float16* p = rowptr + kb + hi * 8;
  Frag16 f;
  f.h[0] = *(const v8h*)(p);
  f.h[1] = *(const v8h*)(p + 16);
  return f.v;
}

// ---------------- utility kernels ----------------
__global__ void zero_b32(uint32_t* p, long long n) {
  long long i = (long long)blockIdx.x * blockDim.x + threadIdx.x;
  long long st = (long long)gridDim.x * blockDim.x;
  for (; i < n; i += st) p[i] = 0u;
}

__global__ void cvt_f16(const float* __restrict__ s, _Float16* __restrict__ d,
                        long long n) {
  long long i = (long long)blockIdx.x * blockDim.x + threadIdx.x;
  long long st = (long long)gridDim.x * blockDim.x;
  for (; i < n; i += st) d[i] = (_Float16)s[i];
}

// ---------------- QKV projection: qkv = x @ w_qkv^T ----------------
// Register-blocked: each wave computes a 64x32 macro-tile (4 M x 2 N tiles),
// reusing A fragments 2x and B fragments 4x (8 WMMAs per 6 fragment loads).
__global__ void __launch_bounds__(256)
qkv_gemm(const _Float16* __restrict__ xh, const _Float16* __restrict__ wh,
         _Float16* __restrict__ qb, _Float16* __restrict__ kbuf,
         _Float16* __restrict__ vT) {
  const int macros = (MT / 4) * (144 / 2);    // 197 * 72
  int macro = blockIdx.x * 8 + (threadIdx.x >> 5);
  if (macro >= macros) return;
  int mm = macro / 72, nn = macro % 72;
  int lane = threadIdx.x & 31, r = lane & 15, hi = lane >> 4;

  const _Float16* a0 = xh + (size_t)(mm * 64 + r) * CH;
  const _Float16* b0 = wh + (size_t)(nn * 32 + r) * CH;

  v8f acc[4][2] = {};
  for (int kb = 0; kb < CH; kb += 32) {
    v16h av[4], bv[2];
#pragma unroll
    for (int mi = 0; mi < 4; ++mi)
      av[mi] = load_frag(a0 + (size_t)mi * 16 * CH, kb, hi);
#pragma unroll
    for (int ni = 0; ni < 2; ++ni)
      bv[ni] = load_frag(b0 + (size_t)ni * 16 * CH, kb, hi);
#pragma unroll
    for (int mi = 0; mi < 4; ++mi)
#pragma unroll
      for (int ni = 0; ni < 2; ++ni)
        acc[mi][ni] = wmma16(av[mi], bv[ni], acc[mi][ni]);
  }

#pragma unroll
  for (int ni = 0; ni < 2; ++ni) {
    int col   = nn * 32 + ni * 16 + r;        // 0..2303
    int which = col / CH;                     // 0=q 1=k 2=v
    int rem   = col % CH;
    int h = rem >> 6, d = rem & 63;
#pragma unroll
    for (int mi = 0; mi < 4; ++mi) {
      Acc8 u; u.v = acc[mi][ni];
      for (int i = 0; i < 8; ++i) {
        int row = mm * 64 + mi * 16 + i + 8 * hi;   // < BN always
        int b_ = row / SEQ, n = row % SEQ;
        size_t bh = (size_t)b_ * HEADS + h;
        if (which == 0)
          qb[(bh * NP + n) * HD + d] = (_Float16)(u.f[i] * 0.125f); // *hd^-0.5
        else if (which == 1)
          kbuf[(bh * NP + n) * HD + d] = (_Float16)u.f[i];
        else
          vT[(bh * HD + d) * NP + n] = (_Float16)u.f[i];
      }
    }
  }
}

// ---------------- attention: one wave per (b, h, q-tile) ----------------
__global__ void __launch_bounds__(32)
attn_kernel(const _Float16* __restrict__ q, const _Float16* __restrict__ k,
            const _Float16* __restrict__ vT, const float* __restrict__ mask,
            _Float16* __restrict__ oh, _Float16* __restrict__ pbuf) {
  __shared__ float    sS[16 * NP];
  __shared__ _Float16 sP[16 * NP];

  int bh = blockIdx.x / 14, qt = blockIdx.x % 14;
  int b_ = bh / HEADS, h = bh % HEADS;
  int lane = threadIdx.x & 31, r = lane & 15, hi = lane >> 4;

  const _Float16* Q = q  + (size_t)bh * NP * HD;
  const _Float16* K = k  + (size_t)bh * NP * HD;
  const _Float16* V = vT + (size_t)bh * HD * NP;
  const _Float16* qrow = Q + (size_t)(qt * 16 + r) * HD;

  // S = (scaled Q) K^T over 14 key tiles, K-dim = 64
  for (int nt = 0; nt < 14; ++nt) {
    const _Float16* krow = K + (size_t)(nt * 16 + r) * HD;
    v8f acc = {};
    for (int kb = 0; kb < HD; kb += 32) {
      v16h a = load_frag(qrow, kb, hi);
      v16h b = load_frag(krow, kb, hi);
      acc = wmma16(a, b, acc);
    }
    Acc8 u; u.v = acc;
    for (int i = 0; i < 8; ++i)
      sS[(i + 8 * hi) * NP + nt * 16 + r] = u.f[i];
  }
  __syncthreads();

  // masked softmax-like normalization, rows handled by lanes 0..15
  if (lane < 16) {
    int m = lane, qr = qt * 16 + m;
    float*    srow = &sS[m * NP];
    _Float16* prow = &sP[m * NP];
    if (qr < SEQ) {
      float mx = -3.4e38f;
      for (int j = 0; j < SEQ; ++j) mx = fmaxf(mx, srow[j]);
      const float* mrow = mask + ((size_t)b_ * SEQ + qr) * SEQ;
      float sum = 0.f;
      for (int j = 0; j < SEQ; ++j) {
        float e = __expf(srow[j] - mx) * mrow[j];
        srow[j] = e;
        sum += e;
      }
      float inv = 1.f / (sum + 1e-6f);
      const float addn = 1e-6f / (float)SEQ;
      for (int j = 0; j < NP; ++j) {
        float p = (j < SEQ) ? (srow[j] + addn) * inv : 0.f;
        prow[j] = (_Float16)p;
      }
    } else {
      for (int j = 0; j < NP; ++j) prow[j] = (_Float16)0.f;
    }
  }
  __syncthreads();

  // deterministic export of normalized probabilities for head reduction
  {
    _Float16* prow0 = pbuf + ((size_t)bh * SEQ) * NP;
    for (int idx = lane; idx < 16 * (NP / 8); idx += 32) {
      int m = idx / (NP / 8), c = idx % (NP / 8);
      int qr = qt * 16 + m;
      if (qr < SEQ)
        *(v8h*)(prow0 + (size_t)qr * NP + c * 8) =
            *(const v8h*)(&sP[m * NP + c * 8]);
    }
  }

  // O = P @ V  (K-dim = NP = 224 = 7*32), B-fragments from V^T rows
  const _Float16* prow_f = &sP[r * NP];
  for (int dt = 0; dt < 4; ++dt) {
    const _Float16* vrow = V + (size_t)(dt * 16 + r) * NP;
    v8f acc = {};
    for (int kb = 0; kb < NP; kb += 32) {
      v16h a = load_frag(prow_f, kb, hi);
      v16h b = load_frag(vrow, kb, hi);
      acc = wmma16(a, b, acc);
    }
    Acc8 u; u.v = acc;
    for (int i = 0; i < 8; ++i) {
      int m = i + 8 * hi, qr = qt * 16 + m;
      if (qr < SEQ)
        oh[((size_t)b_ * SEQ + qr) * CH + h * HD + dt * 16 + r] =
            (_Float16)u.f[i];
    }
  }
}

// ------- deterministic head reduction for cls_attn / patch_attn -------
__global__ void head_reduce(const _Float16* __restrict__ pbuf,
                            float* __restrict__ cls,
                            float* __restrict__ patch) {
  const long long total = (long long)BATCH * SEQ * (SEQ - 1); // (b, qr, j)
  long long i = (long long)blockIdx.x * blockDim.x + threadIdx.x;
  long long st = (long long)gridDim.x * blockDim.x;
  for (; i < total; i += st) {
    int j  = (int)(i % (SEQ - 1));
    long long t = i / (SEQ - 1);
    int qr = (int)(t % SEQ);
    int b_ = (int)(t / SEQ);
    size_t base = (((size_t)b_ * HEADS) * SEQ + qr) * NP + (j + 1);
    float s = 0.f;
    for (int h = 0; h < HEADS; ++h)
      s += (float)pbuf[base + (size_t)h * SEQ * NP];
    s *= (1.f / (float)HEADS);
    if (qr == 0)
      cls[(size_t)b_ * (SEQ - 1) + j] = s;
    else
      patch[((size_t)b_ * (SEQ - 1) + (qr - 1)) * (SEQ - 1) + j] = s;
  }
}

// ---------------- output projection: x_out = oh @ w_proj^T + b ----------------
__global__ void __launch_bounds__(256)
proj_gemm(const _Float16* __restrict__ oh, const _Float16* __restrict__ wh,
          const float* __restrict__ bias, float* __restrict__ out) {
  const int macros = (MT / 4) * (48 / 2);     // 197 * 24
  int macro = blockIdx.x * 8 + (threadIdx.x >> 5);
  if (macro >= macros) return;
  int mm = macro / 24, nn = macro % 24;
  int lane = threadIdx.x & 31, r = lane & 15, hi = lane >> 4;

  const _Float16* a0 = oh + (size_t)(mm * 64 + r) * CH;
  const _Float16* b0 = wh + (size_t)(nn * 32 + r) * CH;

  v8f acc[4][2] = {};
  for (int kb = 0; kb < CH; kb += 32) {
    v16h av[4], bv[2];
#pragma unroll
    for (int mi = 0; mi < 4; ++mi)
      av[mi] = load_frag(a0 + (size_t)mi * 16 * CH, kb, hi);
#pragma unroll
    for (int ni = 0; ni < 2; ++ni)
      bv[ni] = load_frag(b0 + (size_t)ni * 16 * CH, kb, hi);
#pragma unroll
    for (int mi = 0; mi < 4; ++mi)
#pragma unroll
      for (int ni = 0; ni < 2; ++ni)
        acc[mi][ni] = wmma16(av[mi], bv[ni], acc[mi][ni]);
  }

#pragma unroll
  for (int ni = 0; ni < 2; ++ni) {
    int col = nn * 32 + ni * 16 + r;
    float bv = bias[col];
#pragma unroll
    for (int mi = 0; mi < 4; ++mi) {
      Acc8 u; u.v = acc[mi][ni];
      for (int i = 0; i < 8; ++i) {
        int row = mm * 64 + mi * 16 + i + 8 * hi;
        out[(size_t)row * CH + col] = u.f[i] + bv;
      }
    }
  }
}

extern "C" void kernel_launch(void* const* d_in, const int* in_sizes, int n_in,
                              void* d_out, int out_size, void* d_ws,
                              size_t ws_size, hipStream_t stream) {
  (void)in_sizes; (void)n_in; (void)out_size; (void)ws_size;
  const float* x     = (const float*)d_in[0];
  const float* mask  = (const float*)d_in[1];
  const float* wqkv  = (const float*)d_in[2];
  const float* wproj = (const float*)d_in[3];
  const float* bproj = (const float*)d_in[4];

  float* out   = (float*)d_out;
  float* cls   = out + (size_t)BN * CH;                 // 9,682,944
  float* patch = cls + (size_t)BATCH * (SEQ - 1);       // +12,544

  // workspace carve-up (all chunks are multiples of 256 bytes)
  char* ws = (char*)d_ws;
  _Float16* xh     = (_Float16*)ws; ws += (size_t)BN * CH * 2;
  _Float16* wqkvh  = (_Float16*)ws; ws += (size_t)3 * CH * CH * 2;
  _Float16* wprojh = (_Float16*)ws; ws += (size_t)CH * CH * 2;
  _Float16* qb     = (_Float16*)ws; ws += (size_t)BATCH * HEADS * NP * HD * 2;
  _Float16* kb     = (_Float16*)ws; ws += (size_t)BATCH * HEADS * NP * HD * 2;
  _Float16* vT     = (_Float16*)ws; ws += (size_t)BATCH * HEADS * HD * NP * 2;
  _Float16* oh     = (_Float16*)ws; ws += (size_t)BN * CH * 2;
  _Float16* pbuf   = (_Float16*)ws; ws += (size_t)BATCH * HEADS * SEQ * NP * 2;

  // zero padded q/k/vT (contiguous region)
  long long qkv_words = (long long)3 * BATCH * HEADS * NP * HD / 2;
  zero_b32<<<2048, 256, 0, stream>>>((uint32_t*)qb, qkv_words);

  // f32 -> f16 conversions
  cvt_f16<<<2048, 256, 0, stream>>>(x, xh, (long long)BN * CH);
  cvt_f16<<<1024, 256, 0, stream>>>(wqkv, wqkvh, (long long)3 * CH * CH);
  cvt_f16<<<512, 256, 0, stream>>>(wproj, wprojh, (long long)CH * CH);

  // QKV projection: 197*72 macro-tiles, 8 waves (macro-tiles) per block
  qkv_gemm<<<(197 * 72) / 8, 256, 0, stream>>>(xh, wqkvh, qb, kb, vT);

  // attention: one wave per (b,h,q-tile)
  attn_kernel<<<BATCH * HEADS * 14, 32, 0, stream>>>(qb, kb, vT, mask, oh,
                                                     pbuf);

  // deterministic head reduction into cls_attn / patch_attn
  head_reduce<<<2048, 256, 0, stream>>>(pbuf, cls, patch);

  // output projection: 197*24 macro-tiles
  proj_gemm<<<(197 * 24) / 8, 256, 0, stream>>>(oh, wprojh, bproj, out);
}
